// Transformer_31387620999625
// MI455X (gfx1250) — matrix operand order repaired and checked by reading, not compiled, not run on hardware
//
#include <hip/hip_runtime.h>

#define L_ 6
#define NX_ 768
#define H_ 12
#define D_ 64
#define S_ 512
#define V_ 512
#define B_ 4

typedef __attribute__((ext_vector_type(16))) _Float16 v16h;
typedef __attribute__((ext_vector_type(8)))  _Float16 v8h;
typedef __attribute__((ext_vector_type(4)))  _Float16 v4h;
typedef __attribute__((ext_vector_type(8)))  float    v8f;
typedef __attribute__((ext_vector_type(4)))  float    v4f;

// -------- batched-offset helper: off = (z/div)*so + (z%div)*si --------
__device__ __forceinline__ long bmap(int z, int dv, long so, long si) {
    return (long)(z / dv) * so + (long)(z % dv) * si;
}

__device__ __forceinline__ v4h cvt4(v4f v) {
    v4h h;
    h[0] = (_Float16)v[0]; h[1] = (_Float16)v[1];
    h[2] = (_Float16)v[2]; h[3] = (_Float16)v[3];
    return h;
}

// ---------------------------------------------------------------------
// LDS-tiled WMMA GEMM:  C = A * B (+bias) (+resid)
//   A: [M,K] fp32 row-major (lda)
//   B: TRANSB ? [N,K] (ldb) : [K,N] (ldb)
//   C: [M,N] fp32 (ldc)
// Block = 256 threads = 8 waves. Block tile = 128(M) x 64(N) x 32(K).
// Wave w computes rows [mTile+16w, +16), all 64 N cols (4 accumulators),
// reusing one A fragment across 4 WMMAs. A/B tiles staged in LDS as f16.
// Staging is load-all -> convert-all -> store-all so the 6 global_b128
// loads are issued back-to-back (one loadcnt wait) per K step.
// All M multiples of 128, N of 64, K of 32 for this model (no guards).
// ---------------------------------------------------------------------
template <bool TRANSB>
__global__ __launch_bounds__(256)
void gemm_wmma(const float* __restrict__ A, const float* __restrict__ Bm,
               const float* __restrict__ bias, const float* __restrict__ resid,
               float* __restrict__ C,
               int M, int N, int K, int lda, int ldb, int ldc,
               int aDiv, long aSo, long aSi,
               int bDiv, long bSo, long bSi,
               int cDiv, long cSo, long cSi)
{
    __shared__ _Float16 lA[128 * 32];   // [row][k]   8 KB
    __shared__ _Float16 lB[64 * 32];    // [n][k]     4 KB

    const int z = blockIdx.z;
    const float* Ab = A  + bmap(z, aDiv, aSo, aSi);
    const float* Bb = Bm + bmap(z, bDiv, bSo, bSi);
    const long  cOff = bmap(z, cDiv, cSo, cSi);

    const int lane = threadIdx.x & 31;
    const int wave = threadIdx.x >> 5;
    const int col  = lane & 15;   // N-col for B/C/D; M-row within wave tile for A
    const int half = lane >> 4;

    const int mTile = blockIdx.y * 128;
    const int nTile = blockIdx.x * 64;

    // per-thread staging coordinates (fixed across K steps)
    const int tid = (int)threadIdx.x;
    int aR[4], aC[4];
    #pragma unroll
    for (int i = 0; i < 4; ++i) {
        int e4 = tid + i * 256;            // 0..1023 ; 8 float4 per 32-wide row
        aR[i] = e4 >> 3;
        aC[i] = (e4 & 7) << 2;
    }
    int bX[2], bY[2];                      // TRANSB: (n, k4) ; NN: (kk, n4)
    #pragma unroll
    for (int i = 0; i < 2; ++i) {
        int e4 = tid + i * 256;            // 0..511
        if (TRANSB) { bX[i] = e4 >> 3;  bY[i] = (e4 & 7) << 2; }
        else        { bX[i] = e4 >> 4;  bY[i] = (e4 & 15) << 2; }
    }

    v8f acc[4];
    #pragma unroll
    for (int ns = 0; ns < 4; ++ns) acc[ns] = (v8f){0.f,0.f,0.f,0.f,0.f,0.f,0.f,0.f};

    for (int k0 = 0; k0 < K; k0 += 32) {
        // ---- issue ALL global loads for this K step first ----
        v4f va[4], vb[2];
        #pragma unroll
        for (int i = 0; i < 4; ++i)
            va[i] = *(const v4f*)(Ab + (long)(mTile + aR[i]) * lda + k0 + aC[i]);
        #pragma unroll
        for (int i = 0; i < 2; ++i) {
            if (TRANSB)
                vb[i] = *(const v4f*)(Bb + (long)(nTile + bX[i]) * ldb + k0 + bY[i]);
            else
                vb[i] = *(const v4f*)(Bb + (long)(k0 + bX[i]) * ldb + nTile + bY[i]);
        }
        // speculative prefetch of next K tile (global_prefetch_b8)
        if (k0 + 32 < K) {
            __builtin_prefetch(Ab + (long)(mTile + (tid >> 1)) * lda + k0 + 32, 0, 1);
        }
        // ---- convert + store to LDS ----
        #pragma unroll
        for (int i = 0; i < 4; ++i)
            *(v4h*)(&lA[aR[i] * 32 + aC[i]]) = cvt4(va[i]);
        #pragma unroll
        for (int i = 0; i < 2; ++i) {
            if (TRANSB) {
                *(v4h*)(&lB[bX[i] * 32 + bY[i]]) = cvt4(vb[i]);
            } else {
                #pragma unroll
                for (int j = 0; j < 4; ++j)
                    lB[(bY[i] + j) * 32 + bX[i]] = (_Float16)vb[i][j];
            }
        }
        __syncthreads();

        // ---- A fragment for this wave's 16-row block ----
        // a[0..7] = K 8*half+0..7 ; a[8..15] = K 16+8*half+0..7  (ISA layout)
        const int rowA = wave * 16 + col;
        v8h r0 = *(const v8h*)(&lA[rowA * 32 + 8 * half]);
        v8h r1 = *(const v8h*)(&lA[rowA * 32 + 16 + 8 * half]);
        v16h afrag = __builtin_shufflevector(r0, r1,
            0,1,2,3,4,5,6,7,8,9,10,11,12,13,14,15);

        // ---- 4 WMMAs sharing the A fragment ----
        #pragma unroll
        for (int ns = 0; ns < 4; ++ns) {
            // b[0..7] = K 8*half+0..7 ; b[8..15] = K 16+8*half+0..7 of column ns*16+col
            v8h q0 = *(const v8h*)(&lB[(ns * 16 + col) * 32 + 8 * half]);
            v8h q1 = *(const v8h*)(&lB[(ns * 16 + col) * 32 + 16 + 8 * half]);
            v16h bfrag = __builtin_shufflevector(q0, q1,
                0,1,2,3,4,5,6,7,8,9,10,11,12,13,14,15);
            acc[ns] = __builtin_amdgcn_wmma_f32_16x16x32_f16(
                false, afrag, false, bfrag, (short)0, acc[ns], false, false);
        }
        __syncthreads();
    }

    // ---- store: D layout VGPR r -> M = r + 8*half, N = col ----
    const int gmBase = mTile + wave * 16;
    #pragma unroll
    for (int ns = 0; ns < 4; ++ns) {
        const int gn = nTile + ns * 16 + col;
        const float bv = bias ? bias[gn] : 0.f;
        #pragma unroll
        for (int r = 0; r < 8; ++r) {
            int gm = gmBase + r + 8 * half;
            long idx = cOff + (long)gm * ldc + gn;
            float val = acc[ns][r] + bv;
            if (resid) val += resid[idx];
            C[idx] = val;
        }
    }
}

// ------------------------- elementwise kernels -------------------------

__global__ void embed_kernel(const int* __restrict__ idx,
                             const float* __restrict__ wte,
                             float* __restrict__ h)
{
    long t = (long)blockIdx.x * blockDim.x + threadIdx.x;   // B*S*NX
    int  e  = (int)(t % NX_);
    long bs = t / NX_;
    h[t] = wte[(long)idx[bs] * NX_ + e];
}

__global__ void ln_kernel(const float* __restrict__ x, const float* __restrict__ g,
                          const float* __restrict__ bt, float* __restrict__ y)
{
    __shared__ float red[256];
    const int row = blockIdx.x;
    const float* xr = x + (long)row * NX_;
    float s = 0.f;
    for (int i = threadIdx.x; i < NX_; i += 256) s += xr[i];
    red[threadIdx.x] = s; __syncthreads();
    for (int w = 128; w > 0; w >>= 1) {
        if ((int)threadIdx.x < w) red[threadIdx.x] += red[threadIdx.x + w];
        __syncthreads();
    }
    const float mean = red[0] / NX_;
    __syncthreads();
    float v = 0.f;
    for (int i = threadIdx.x; i < NX_; i += 256) { float d = xr[i] - mean; v += d * d; }
    red[threadIdx.x] = v; __syncthreads();
    for (int w = 128; w > 0; w >>= 1) {
        if ((int)threadIdx.x < w) red[threadIdx.x] += red[threadIdx.x + w];
        __syncthreads();
    }
    const float rstd = rsqrtf(red[0] / NX_ + 1e-5f);
    float* yr = y + (long)row * NX_;
    for (int i = threadIdx.x; i < NX_; i += 256)
        yr[i] = (xr[i] - mean) * rstd * g[i] + bt[i];
}

// split qkv [B,S,3NX] -> q [B,H,S,D] and present [B,L,2,H,S,D] (k,v)
__global__ void qkv_split_kernel(const float* __restrict__ qkv,
                                 float* __restrict__ qb,
                                 float* __restrict__ present, int l)
{
    long t = (long)blockIdx.x * blockDim.x + threadIdx.x;   // B*S*NX
    int  d  = (int)(t % D_);  long t2 = t / D_;
    int  hh = (int)(t2 % H_); long t3 = t2 / H_;
    int  s  = (int)(t3 % S_); int  b  = (int)(t3 / S_);
    const float* row = qkv + (long)(b * S_ + s) * (3 * NX_);
    const int cb = hh * D_ + d;
    qb[(((long)(b * H_ + hh)) * S_ + s) * D_ + d] = row[cb];
    const long pk = ((((long)((b * L_ + l) * 2 + 0) * H_ + hh) * S_ + s)) * D_ + d;
    const long pv = ((((long)((b * L_ + l) * 2 + 1) * H_ + hh) * S_ + s)) * D_ + d;
    present[pk] = row[NX_ + cb];
    present[pv] = row[2 * NX_ + cb];
}

// scores[bh,q,:] = softmax( (scores + skew(rel)) / sqrt(D) , causal )
__global__ void attn_softmax_kernel(float* __restrict__ scores,
                                    const float* __restrict__ rel)
{
    __shared__ float red[256];
    const int q  = blockIdx.x;
    const int bh = blockIdx.y;
    float* wr = scores + ((long)bh * S_ + q) * S_;
    const float* rm = rel + (long)bh * S_ * S_;
    const float scale = 0.125f;                 // 1/sqrt(64)
    float loc[2];
    float mx = -3.0e38f;
    for (int k = threadIdx.x, j = 0; k < S_; k += 256, ++j) {
        long f = (long)(q + 1) * S_ + k;        // Music-Transformer skew trick
        int  r = (int)(f / (S_ + 1));
        int  c = (int)(f % (S_ + 1));
        float rv = (c == 0) ? 0.f : rm[(long)r * S_ + (c - 1)];
        float val = (wr[k] + rv) * scale;
        if (k > q) val = -1e30f;
        loc[j] = val;
        mx = fmaxf(mx, val);
    }
    red[threadIdx.x] = mx; __syncthreads();
    for (int w = 128; w > 0; w >>= 1) {
        if ((int)threadIdx.x < w) red[threadIdx.x] = fmaxf(red[threadIdx.x], red[threadIdx.x + w]);
        __syncthreads();
    }
    mx = red[0]; __syncthreads();
    float sum = 0.f;
    for (int k = threadIdx.x, j = 0; k < S_; k += 256, ++j) {
        loc[j] = __expf(loc[j] - mx);
        sum += loc[j];
    }
    red[threadIdx.x] = sum; __syncthreads();
    for (int w = 128; w > 0; w >>= 1) {
        if ((int)threadIdx.x < w) red[threadIdx.x] += red[threadIdx.x + w];
        __syncthreads();
    }
    const float inv = 1.f / red[0];
    for (int k = threadIdx.x, j = 0; k < S_; k += 256, ++j)
        wr[k] = (k > q) ? 0.f : loc[j] * inv;
}

__global__ void gelu_kernel(float* __restrict__ x, long n)
{
    long t = (long)blockIdx.x * blockDim.x + threadIdx.x;
    if (t >= n) return;
    float v = x[t];
    const float c = 0.7978845608028654f;        // sqrt(2/pi)
    float u = c * (v + 0.044715f * v * v * v);
    x[t] = 0.5f * v * (1.f + tanhf(u));
}

// ----------------------------- host side ------------------------------

extern "C" void kernel_launch(void* const* d_in, const int* in_sizes, int n_in,
                              void* d_out, int out_size, void* d_ws, size_t ws_size,
                              hipStream_t stream)
{
    (void)in_sizes; (void)n_in; (void)out_size; (void)ws_size;
    const int*   inputs  = (const int*)  d_in[0];
    const float* wte     = (const float*)d_in[1];
    const float* ln1_g   = (const float*)d_in[2];
    const float* ln1_b   = (const float*)d_in[3];
    const float* attn_w  = (const float*)d_in[4];
    const float* attn_b  = (const float*)d_in[5];
    const float* E       = (const float*)d_in[6];
    const float* proj_w  = (const float*)d_in[7];
    const float* proj_b  = (const float*)d_in[8];
    const float* ln2_g   = (const float*)d_in[9];
    const float* ln2_b   = (const float*)d_in[10];
    const float* fc_w    = (const float*)d_in[11];
    const float* fc_b    = (const float*)d_in[12];
    const float* mproj_w = (const float*)d_in[13];
    const float* mproj_b = (const float*)d_in[14];
    const float* lnf_g   = (const float*)d_in[15];
    const float* lnf_b   = (const float*)d_in[16];

    float* logits  = (float*)d_out;                          // [B,S,V]
    float* present = logits + (long)B_ * S_ * V_;            // [B,L,2,H,S,D]

    const long nTok = (long)B_ * S_;                          // 2048
    float* hbuf   = (float*)d_ws;                             // [nTok,NX]
    float* xn     = hbuf + nTok * NX_;                        // [nTok,NX] (also attn-out)
    float* big    = xn + nTok * NX_;                          // [nTok,4*NX] qkv/fc shared
    float* qb     = big + nTok * 4 * NX_;                     // [B,H,S,D]
    float* scores = qb + (long)B_ * H_ * S_ * D_;             // [B,H,S,S]
    float* rel    = scores + (long)B_ * H_ * S_ * S_;         // [B,H,S,S]

    const long presLayerKV = (long)H_ * S_ * D_;              // one (l,kv) slab
    const long presBStride = (long)L_ * 2 * H_ * S_ * D_;     // batch b stride
    const int  BH = B_ * H_;

    // embedding gather
    embed_kernel<<<dim3((unsigned)(nTok * NX_ / 256)), dim3(256), 0, stream>>>(inputs, wte, hbuf);

    for (int l = 0; l < L_; ++l) {
        // ln1
        ln_kernel<<<dim3((unsigned)nTok), dim3(256), 0, stream>>>(hbuf, ln1_g + l * NX_, ln1_b + l * NX_, xn);

        // qkv = xn @ attn_w[l] + attn_b[l]   [2048 x 2304]
        gemm_wmma<false><<<dim3(3 * NX_ / 64, (unsigned)(nTok / 128), 1), dim3(256), 0, stream>>>(
            xn, attn_w + (long)l * NX_ * 3 * NX_, attn_b + (long)l * 3 * NX_, nullptr, big,
            (int)nTok, 3 * NX_, NX_, NX_, 3 * NX_, 3 * NX_,
            1, 0, 0, 1, 0, 0, 1, 0, 0);

        // split into q-buffer and present (k,v)
        qkv_split_kernel<<<dim3((unsigned)(nTok * NX_ / 256)), dim3(256), 0, stream>>>(big, qb, present, l);

        // scores[bh] = q[bh] @ k[bh]^T   (k read from present)
        gemm_wmma<true><<<dim3(S_ / 64, S_ / 128, BH), dim3(256), 0, stream>>>(
            qb, present + (long)(l * 2 + 0) * presLayerKV, nullptr, nullptr, scores,
            S_, S_, D_, D_, D_, S_,
            1, (long)S_ * D_, 0,
            H_, presBStride, (long)S_ * D_,
            1, (long)S_ * S_, 0);

        // rel[bh] = q[bh] @ E[l, h]^T   (E shared across batch b -> z%H mapping)
        gemm_wmma<true><<<dim3(S_ / 64, S_ / 128, BH), dim3(256), 0, stream>>>(
            qb, E + (long)l * H_ * S_ * D_, nullptr, nullptr, rel,
            S_, S_, D_, D_, D_, S_,
            1, (long)S_ * D_, 0,
            H_, 0, (long)S_ * D_,
            1, (long)S_ * S_, 0);

        // softmax with skewed relative bias + causal mask (in place on scores)
        attn_softmax_kernel<<<dim3(S_, BH), dim3(256), 0, stream>>>(scores, rel);

        // attn-out(merged heads) = P @ v ; C written directly as [B,S,H*D] into xn
        gemm_wmma<false><<<dim3(D_ / 64, S_ / 128, BH), dim3(256), 0, stream>>>(
            scores, present + (long)(l * 2 + 1) * presLayerKV, nullptr, nullptr, xn,
            S_, D_, S_, S_, D_, NX_,
            1, (long)S_ * S_, 0,
            H_, presBStride, (long)S_ * D_,
            H_, (long)S_ * NX_, (long)D_);

        // h += attn_out @ proj_w[l] + proj_b[l]
        gemm_wmma<false><<<dim3(NX_ / 64, (unsigned)(nTok / 128), 1), dim3(256), 0, stream>>>(
            xn, proj_w + (long)l * NX_ * NX_, proj_b + (long)l * NX_, hbuf, hbuf,
            (int)nTok, NX_, NX_, NX_, NX_, NX_,
            1, 0, 0, 1, 0, 0, 1, 0, 0);

        // ln2
        ln_kernel<<<dim3((unsigned)nTok), dim3(256), 0, stream>>>(hbuf, ln2_g + l * NX_, ln2_b + l * NX_, xn);

        // fc = xn @ fc_w[l] + fc_b[l]  [2048 x 3072]
        gemm_wmma<false><<<dim3(4 * NX_ / 64, (unsigned)(nTok / 128), 1), dim3(256), 0, stream>>>(
            xn, fc_w + (long)l * NX_ * 4 * NX_, fc_b + (long)l * 4 * NX_, nullptr, big,
            (int)nTok, 4 * NX_, NX_, NX_, 4 * NX_, 4 * NX_,
            1, 0, 0, 1, 0, 0, 1, 0, 0);

        // gelu in place
        gelu_kernel<<<dim3((unsigned)(nTok * 4 * NX_ / 256)), dim3(256), 0, stream>>>(big, nTok * 4 * NX_);

        // h += gelu(fc) @ mproj_w[l] + mproj_b[l]
        gemm_wmma<false><<<dim3(NX_ / 64, (unsigned)(nTok / 128), 1), dim3(256), 0, stream>>>(
            big, mproj_w + (long)l * 4 * NX_ * NX_, mproj_b + (long)l * NX_, hbuf, hbuf,
            (int)nTok, NX_, 4 * NX_, 4 * NX_, NX_, NX_,
            1, 0, 0, 1, 0, 0, 1, 0, 0);
    }

    // final LN + tied-embedding head: logits = ln_f(h) @ wte^T
    ln_kernel<<<dim3((unsigned)nTok), dim3(256), 0, stream>>>(hbuf, lnf_g, lnf_b, xn);
    gemm_wmma<true><<<dim3(V_ / 64, (unsigned)(nTok / 128), 1), dim3(256), 0, stream>>>(
        xn, wte, nullptr, nullptr, logits,
        (int)nTok, V_, NX_, NX_, NX_, V_,
        1, 0, 0, 1, 0, 0, 1, 0, 0);
}
